// AttnDecoder_SEQ_34359738368619
// MI455X (gfx1250) — matrix operand order repaired and checked by reading, not compiled, not run on hardware
//
#include <hip/hip_runtime.h>
#include <cstdint>
#include <cstddef>

#define V_SIZE 32000
#define E_SIZE 512
#define H_SIZE 512
#define L_SIZE 2
#define MAXLEN_ 99
#define B_SIZE 4096

typedef __attribute__((ext_vector_type(16))) __bf16 v16bf;
typedef __attribute__((ext_vector_type(8)))  float v8f;
typedef __attribute__((ext_vector_type(4)))  int   v4i;

union FragBF { v16bf v; unsigned u[8]; };

#define AS1 __attribute__((address_space(1)))
#define AS3 __attribute__((address_space(3)))

#if __has_builtin(__builtin_amdgcn_global_load_async_to_lds_b128) && \
    __has_builtin(__builtin_amdgcn_s_wait_asynccnt)
#define HAVE_ASYNC_LDS 1
#else
#define HAVE_ASYNC_LDS 0
#endif

__device__ __forceinline__ void async_copy16(const unsigned short* g, unsigned short* l) {
#if HAVE_ASYNC_LDS
  __builtin_amdgcn_global_load_async_to_lds_b128((AS1 v4i*)g, (AS3 v4i*)l, 0, 0);
#else
  uint4 v = *reinterpret_cast<const uint4*>(g);
  uint2* d = reinterpret_cast<uint2*>(l);
  d[0] = make_uint2(v.x, v.y);
  d[1] = make_uint2(v.z, v.w);
#endif
}

template<int N>
__device__ __forceinline__ void wait_async() {
#if HAVE_ASYNC_LDS
  __builtin_amdgcn_s_wait_asynccnt(N);
#endif
}

__device__ __forceinline__ unsigned short f2bf(float f) {
  unsigned x = __float_as_uint(f);
  unsigned r = x + 0x7fffu + ((x >> 16) & 1u);   // round-to-nearest-even
  return (unsigned short)(r >> 16);
}

// ---------------- fp32 -> bf16 bulk convert ----------------
__global__ void k_f32_to_bf16(const float* __restrict__ src,
                              unsigned short* __restrict__ dst, int n) {
  int i = blockIdx.x * blockDim.x + threadIdx.x;
  if (i < n) dst[i] = f2bf(src[i]);
}

// ---------------- embedding gather ----------------
__global__ void k_gather_emb(const int* __restrict__ idx,
                             const float* __restrict__ emb_tab,
                             float* __restrict__ emb_f32,
                             unsigned short* __restrict__ xcat) {
  int g = blockIdx.x * blockDim.x + threadIdx.x;  // 0..B*H-1
  int b = g >> 9;
  int e = g & 511;
  float v = emb_tab[(size_t)idx[b] * E_SIZE + e];
  emb_f32[g] = v;
  xcat[(size_t)b * 1024 + e] = f2bf(v);           // first half of concat(emb, attn)
}

// ---------------- attention logits + softmax over 99 ----------------
__global__ __launch_bounds__(128)
void k_attn_softmax(const float* __restrict__ emb,
                    const float* __restrict__ hidden0,
                    const float* __restrict__ attn_W,
                    const float* __restrict__ attn_b,
                    float* __restrict__ attn_out) {
  __shared__ float q[1024];
  __shared__ float lg[128];
  __shared__ float red[2];
  int b = blockIdx.x, t = threadIdx.x;
  for (int j = t; j < 512; j += 128) {
    q[j]       = emb[(size_t)b * 512 + j];
    q[512 + j] = hidden0[(size_t)b * 512 + j];
  }
  __syncthreads();
  float lv = -1e30f;
  if (t < MAXLEN_) {
    const float* wr = attn_W + (size_t)t * 1024;
    float acc = attn_b[t];
    for (int j = 0; j < 1024; ++j) acc += wr[j] * q[j];
    lv = acc;
  }
  lg[t] = lv;
  __syncthreads();
  if (t == 0) {
    float m = -1e30f;
    for (int j = 0; j < MAXLEN_; ++j) m = fmaxf(m, lg[j]);
    red[0] = m;
  }
  __syncthreads();
  float ev = 0.f;
  if (t < MAXLEN_) { ev = expf(lv - red[0]); lg[t] = ev; }
  __syncthreads();
  if (t == 0) {
    float s = 0.f;
    for (int j = 0; j < MAXLEN_; ++j) s += lg[j];
    red[1] = 1.f / s;
  }
  __syncthreads();
  if (t < MAXLEN_) attn_out[(size_t)b * MAXLEN_ + t] = ev * red[1];
}

// ---------------- attn_applied = w @ encoder_outputs (per row) ----------------
__global__ __launch_bounds__(256)
void k_attn_apply(const float* __restrict__ attn_w,
                  const float* __restrict__ enc,
                  unsigned short* __restrict__ xcat) {
  __shared__ float w[MAXLEN_];
  int b = blockIdx.x, t = threadIdx.x;
  if (t < MAXLEN_) w[t] = attn_w[(size_t)b * MAXLEN_ + t];
  __syncthreads();
  float a0 = 0.f, a1 = 0.f;
  const float* ebase = enc + (size_t)b * MAXLEN_ * H_SIZE;
  for (int l = 0; l < MAXLEN_; ++l) {
    float wl = w[l];
    const float* row = ebase + (size_t)l * H_SIZE;
    if (l + 8 < MAXLEN_) __builtin_prefetch(ebase + (size_t)(l + 8) * H_SIZE + t, 0, 0);
    a0 += wl * row[t];
    a1 += wl * row[t + 256];
  }
  xcat[(size_t)b * 1024 + 512 + t]       = f2bf(a0);
  xcat[(size_t)b * 1024 + 512 + t + 256] = f2bf(a1);
}

// ---------------- WMMA bf16 GEMM:  C[M,N] = act(A[M,K] @ W[N,K]^T + bias) ----------------
#define BM 128
#define BN 128
#define BK 32
#define LSTR 40   // BK + 8 pad: 80B row stride -> 16B-aligned async stores, conflict-free frag reads

template<bool RELU, bool OUTF, bool OUTB>
__global__ __launch_bounds__(256)
void k_gemm_bf16(const unsigned short* __restrict__ A,
                 const unsigned short* __restrict__ W,
                 const float* __restrict__ bias,
                 float* __restrict__ Cf,
                 unsigned short* __restrict__ Cb,
                 int M, int N, int K, int ldc) {
  __shared__ __align__(16) unsigned short As[2][BM * LSTR];
  __shared__ __align__(16) unsigned short Bs[2][BN * LSTR];
  const int tid  = threadIdx.x;
  const int lane = tid & 31;
  const int wave = tid >> 5;
  const int wm   = wave & 3;     // 4 wave-rows  -> 32 M each
  const int wn   = wave >> 2;    // 2 wave-cols  -> 64 N each
  const int l16  = lane & 15;
  const int half = lane >> 4;
  const int bm0  = blockIdx.y * BM;
  const int bn0  = blockIdx.x * BN;

  v8f zero = {0.f,0.f,0.f,0.f,0.f,0.f,0.f,0.f};
  v8f acc[2][4];
  for (int mi = 0; mi < 2; ++mi)
    for (int ni = 0; ni < 4; ++ni) acc[mi][ni] = zero;

  const int lrow0 = tid >> 2;          // 0..63
  const int lcol  = (tid & 3) * 8;     // 0,8,16,24 (bf16 elems)
  const unsigned short* gA = A + (size_t)(bm0 + lrow0) * K + lcol;
  const unsigned short* gB = W + (size_t)(bn0 + lrow0) * K + lcol;
  const size_t rowStep = (size_t)64 * K;

  // Stage one 128x32 A tile + 128x32 B tile: 4 async B128 ops per wave.
  auto stage = [&](int buf, int k0) {
    #pragma unroll
    for (int it = 0; it < 2; ++it) {
      async_copy16(gA + (size_t)it * rowStep + k0, &As[buf][(lrow0 + it * 64) * LSTR + lcol]);
      async_copy16(gB + (size_t)it * rowStep + k0, &Bs[buf][(lrow0 + it * 64) * LSTR + lcol]);
    }
  };

  const int nK = K / BK;
  stage(0, 0);
  for (int kt = 0; kt < nK; ++kt) {
    const int buf = kt & 1;
    if (kt + 1 < nK) {
      stage(buf ^ 1, (kt + 1) * BK);   // prefetch next tile into other buffer
      wait_async<4>();                 // in-order completion => current tile resident
    } else {
      wait_async<0>();
    }
    __syncthreads();

    FragBF afr[2], bfr[4];
    #pragma unroll
    for (int mi = 0; mi < 2; ++mi) {
      const unsigned short* ar = &As[buf][(wm * 32 + mi * 16 + l16) * LSTR];
      #pragma unroll
      for (int j = 0; j < 8; ++j) {           // 16-bit A layout: K = 2j + 16*half (matches B)
        int k = 2 * j + 16 * half;
        afr[mi].u[j] = *reinterpret_cast<const unsigned*>(&ar[k]);
      }
    }
    #pragma unroll
    for (int ni = 0; ni < 4; ++ni) {
      const unsigned short* br = &Bs[buf][(wn * 64 + ni * 16 + l16) * LSTR];
      #pragma unroll
      for (int j = 0; j < 8; ++j) {           // B layout used here (same as A): K = 2j + 16*half
        int k = 2 * j + 16 * half;
        bfr[ni].u[j] = *reinterpret_cast<const unsigned*>(&br[k]);
      }
    }
    #pragma unroll
    for (int mi = 0; mi < 2; ++mi)
      #pragma unroll
      for (int ni = 0; ni < 4; ++ni)
        acc[mi][ni] = __builtin_amdgcn_wmma_f32_16x16x32_bf16(
            false, afr[mi].v, false, bfr[ni].v, (short)0, acc[mi][ni], false, false);
    __syncthreads();   // everyone done reading buf before it is refilled next iteration
  }

  #pragma unroll
  for (int ni = 0; ni < 4; ++ni) {
    int n = bn0 + wn * 64 + ni * 16 + l16;
    float bv = bias[n];
    #pragma unroll
    for (int mi = 0; mi < 2; ++mi) {
      #pragma unroll
      for (int i = 0; i < 8; ++i) {           // C layout: VGPR i -> row i + 8*half
        int m = bm0 + wm * 32 + mi * 16 + 8 * half + i;
        float v = acc[mi][ni][i] + bv;
        if (RELU) v = fmaxf(v, 0.f);
        size_t off = (size_t)m * ldc + n;
        if (OUTF) Cf[off] = v;
        if (OUTB) Cb[off] = f2bf(v);
      }
    }
  }
}

// ---------------- GRU gate fusion ----------------
__global__ void k_gru(const float* __restrict__ gi, const float* __restrict__ gh,
                      const float* __restrict__ hprev,
                      float* __restrict__ hnew_f, unsigned short* __restrict__ hnew_b) {
  int g = blockIdx.x * blockDim.x + threadIdx.x;  // B*H
  int b = g >> 9, h = g & 511;
  size_t base = (size_t)b * 1536;
  float i_r = gi[base + h], i_z = gi[base + 512 + h], i_n = gi[base + 1024 + h];
  float h_r = gh[base + h], h_z = gh[base + 512 + h], h_n = gh[base + 1024 + h];
  float r = 1.f / (1.f + expf(-(i_r + h_r)));
  float z = 1.f / (1.f + expf(-(i_z + h_z)));
  float n = tanhf(i_n + r * h_n);
  float hp = hprev[g];
  float hn = (1.f - z) * n + z * hp;
  hnew_f[g] = hn;
  hnew_b[g] = f2bf(hn);
}

// ---------------- row softmax over V=32000 (2-pass, online max/sum) ----------------
__global__ __launch_bounds__(256)
void k_softmax_rows(float* __restrict__ logits, int ncols) {
  __shared__ float sm[256], ss[256];
  int b = blockIdx.x, t = threadIdx.x;
  float* row = logits + (size_t)b * ncols;
  float m = -3.4e38f, s = 0.f;
  for (int i = t; i < ncols; i += 256) {
    float x = row[i];
    if (x > m) { s = s * expf(m - x) + 1.f; m = x; }
    else       { s += expf(x - m); }
  }
  sm[t] = m; ss[t] = s;
  __syncthreads();
  for (int o = 128; o > 0; o >>= 1) {
    if (t < o) {
      float m2 = sm[t + o], s2 = ss[t + o];
      float M = fmaxf(sm[t], m2);
      ss[t] = ss[t] * expf(sm[t] - M) + s2 * expf(m2 - M);
      sm[t] = M;
    }
    __syncthreads();
  }
  float M = sm[0], inv = 1.f / ss[0];
  for (int i = t; i < ncols; i += 256) row[i] = expf(row[i] - M) * inv;
}

// ---------------- launcher ----------------
extern "C" void kernel_launch(void* const* d_in, const int* in_sizes, int n_in,
                              void* d_out, int out_size, void* d_ws, size_t ws_size,
                              hipStream_t stream) {
  (void)in_sizes; (void)n_in; (void)out_size; (void)ws_size;
  const int*   inputs  = (const int*)  d_in[0];
  const float* hidden  = (const float*)d_in[1];
  const float* enc     = (const float*)d_in[2];
  const float* emb_tab = (const float*)d_in[3];
  const float* attn_W  = (const float*)d_in[4];
  const float* attn_b  = (const float*)d_in[5];
  const float* comb_W  = (const float*)d_in[6];
  const float* comb_b  = (const float*)d_in[7];
  const float* W_ih    = (const float*)d_in[8];
  const float* W_hh    = (const float*)d_in[9];
  const float* b_ih    = (const float*)d_in[10];
  const float* b_hh    = (const float*)d_in[11];
  const float* out_W   = (const float*)d_in[12];
  const float* out_b   = (const float*)d_in[13];

  float* out      = (float*)d_out;
  float* pred     = out;                                        // B*V
  float* hid_new  = out + (size_t)B_SIZE * V_SIZE;              // L*B*H
  float* attn_out = hid_new + (size_t)L_SIZE * B_SIZE * H_SIZE; // B*99

  char* ws = (char*)d_ws;
  size_t off = 0;
  auto alloc = [&](size_t bytes) {
    void* p = ws + off;
    off = (off + bytes + 255) & ~(size_t)255;
    return p;
  };
  float*          emb_f32 = (float*)         alloc((size_t)B_SIZE * H_SIZE * 4);
  unsigned short* xcat    = (unsigned short*)alloc((size_t)B_SIZE * 1024 * 2);
  unsigned short* x_b     = (unsigned short*)alloc((size_t)B_SIZE * H_SIZE * 2);
  unsigned short* h_b     = (unsigned short*)alloc((size_t)L_SIZE * B_SIZE * H_SIZE * 2);
  unsigned short* hn_b    = (unsigned short*)alloc((size_t)L_SIZE * B_SIZE * H_SIZE * 2);
  float*          gi      = (float*)         alloc((size_t)B_SIZE * 1536 * 4);
  float*          gh      = (float*)         alloc((size_t)B_SIZE * 1536 * 4);
  unsigned short* cW_b    = (unsigned short*)alloc((size_t)512 * 1024 * 2);
  unsigned short* Wih_b   = (unsigned short*)alloc((size_t)L_SIZE * 1536 * 512 * 2);
  unsigned short* Whh_b   = (unsigned short*)alloc((size_t)L_SIZE * 1536 * 512 * 2);
  unsigned short* oW_b    = (unsigned short*)alloc((size_t)V_SIZE * 512 * 2);

  auto cvt = [&](const float* s, unsigned short* d, size_t n) {
    k_f32_to_bf16<<<dim3((unsigned)((n + 255) / 256)), 256, 0, stream>>>(s, d, (int)n);
  };
  cvt(comb_W, cW_b, (size_t)512 * 1024);
  cvt(W_ih,  Wih_b, (size_t)L_SIZE * 1536 * 512);
  cvt(W_hh,  Whh_b, (size_t)L_SIZE * 1536 * 512);
  cvt(out_W, oW_b,  (size_t)V_SIZE * 512);
  cvt(hidden, h_b,  (size_t)L_SIZE * B_SIZE * H_SIZE);

  k_gather_emb<<<(B_SIZE * H_SIZE) / 256, 256, 0, stream>>>(inputs, emb_tab, emb_f32, xcat);
  k_attn_softmax<<<B_SIZE, 128, 0, stream>>>(emb_f32, hidden, attn_W, attn_b, attn_out);
  k_attn_apply<<<B_SIZE, 256, 0, stream>>>(attn_out, enc, xcat);

  // combine: x = relu(concat(emb,attn) @ combine_W^T + b) -> bf16
  k_gemm_bf16<true, false, true><<<dim3(512 / BN, B_SIZE / BM), 256, 0, stream>>>(
      xcat, cW_b, comb_b, nullptr, x_b, B_SIZE, 512, 1024, 512);
  // GRU layer 0
  k_gemm_bf16<false, true, false><<<dim3(1536 / BN, B_SIZE / BM), 256, 0, stream>>>(
      x_b, Wih_b, b_ih, gi, nullptr, B_SIZE, 1536, 512, 1536);
  k_gemm_bf16<false, true, false><<<dim3(1536 / BN, B_SIZE / BM), 256, 0, stream>>>(
      h_b, Whh_b, b_hh, gh, nullptr, B_SIZE, 1536, 512, 1536);
  k_gru<<<(B_SIZE * H_SIZE) / 256, 256, 0, stream>>>(gi, gh, hidden, hid_new, hn_b);
  // GRU layer 1
  k_gemm_bf16<false, true, false><<<dim3(1536 / BN, B_SIZE / BM), 256, 0, stream>>>(
      hn_b, Wih_b + (size_t)1536 * 512, b_ih + 1536, gi, nullptr, B_SIZE, 1536, 512, 1536);
  k_gemm_bf16<false, true, false><<<dim3(1536 / BN, B_SIZE / BM), 256, 0, stream>>>(
      h_b + (size_t)B_SIZE * H_SIZE, Whh_b + (size_t)1536 * 512, b_hh + 1536,
      gh, nullptr, B_SIZE, 1536, 512, 1536);
  k_gru<<<(B_SIZE * H_SIZE) / 256, 256, 0, stream>>>(
      gi, gh, hidden + (size_t)B_SIZE * H_SIZE,
      hid_new + (size_t)B_SIZE * H_SIZE, hn_b + (size_t)B_SIZE * H_SIZE);
  // output projection: logits into pred region, then in-place softmax
  k_gemm_bf16<false, true, false><<<dim3(V_SIZE / BN, B_SIZE / BM), 256, 0, stream>>>(
      hn_b + (size_t)B_SIZE * H_SIZE, oW_b, out_b, pred, nullptr, B_SIZE, V_SIZE, 512, V_SIZE);
  k_softmax_rows<<<B_SIZE, 256, 0, stream>>>(pred, V_SIZE);
}
